// UCL_49417893708249
// MI455X (gfx1250) — compile-verified, matrix operand order checked by reading, not run on hardware
//
#include <hip/hip_runtime.h>
#include <stdint.h>

// ---------------------------------------------------------------------------
// Problem constants (from reference)
// ---------------------------------------------------------------------------
#define ROWS     16384
#define IN_FEA   512
#define OUT_FEA  128
#define HALF_N   8192           // ROWS/2
#define ROW_TILES_FULL  (ROWS/16)    // 1024
#define ROW_TILES_HALF  (HALF_N/16)  // 512
#define E2       7.3890560989306495f // exp(2) == diag(refl) after l2-norm

typedef __attribute__((ext_vector_type(16))) __bf16 bf16x16;
typedef __attribute__((ext_vector_type(8)))  float  f32x8;
typedef __attribute__((ext_vector_type(4)))  unsigned int u32x4;
typedef __attribute__((ext_vector_type(8)))  int   i32x8;
typedef __attribute__((ext_vector_type(4)))  int   i32x4;

union FragB {            // 16x32 bf16 A-frag / 32x16 bf16 B-frag (8 VGPRs)
    bf16x16  v;
    uint32_t u[8];
    uint4    q[2];
};
union AccF {             // 16x16 f32 C/D frag (8 VGPRs)
    f32x8 v;
    float f[8];
};

// Native bf16 converts (RNE) -> v_cvt_*bf16* on gfx1250 instead of ALU chains.
static __device__ inline uint16_t f2bf(float a) {
    union { __bf16 h; uint16_t u; } r;
    r.h = (__bf16)a;
    return r.u;
}
static __device__ inline uint32_t pack2_bf16(float a, float b) {
    union { __bf16 h[2]; uint32_t u; } r;
    r.h[0] = (__bf16)a;
    r.h[1] = (__bf16)b;
    return r.u;
}

static __device__ inline f32x8 wmma_bf16(const FragB& a, const FragB& b, const f32x8& c) {
    return __builtin_amdgcn_wmma_f32_16x16x32_bf16(false, a.v, false, b.v,
                                                   (short)0, c, false, false);
}

// ---- CDNA5 async memory helpers -------------------------------------------
// Generic pointers to __shared__ carry the LDS byte offset in their low 32
// bits (ISA 10.2).
static __device__ inline uint32_t lds_off(const void* p) {
    return (uint32_t)(uintptr_t)p;
}
// GLOBAL_LOAD_ASYNC_TO_LDS_B128: per-lane 16B global->LDS copy, ASYNCcnt.
static __device__ inline void async_ld_b128(uint32_t ldsa, const void* gaddr) {
    asm volatile("global_load_async_to_lds_b128 %0, %1, off"
                 :: "v"(ldsa), "v"(gaddr) : "memory");
}
static __device__ inline void wait_async0() {
    asm volatile("s_wait_asynccnt 0x0" ::: "memory");
}
static __device__ inline void wait_async1() {
    asm volatile("s_wait_asynccnt 0x1" ::: "memory");
}

// A/B fragment loader for bf16 row-major data (global or LDS).
// Layout (ISA 7.12.2, 16-bit A 16x32): lane = 16*h + m ; VGPR v holds K pair
// k = 8*h + (v<4?0:16) + 2*(v%4)  ->  q[0] = 8 bf16 at kb, q[1] = 8 bf16 at kb+16.
// The same per-lane addressing serves B = Y^T (column n of B == row n of Y).
static __device__ inline FragB load_frag_bf16(const uint16_t* base, int row,
                                              int stride, int kb) {
    FragB f;
    const uint16_t* p = base + (size_t)row * stride + kb;
    f.q[0] = *reinterpret_cast<const uint4*>(p);
    f.q[1] = *reinterpret_cast<const uint4*>(p + 16);
    return f;
}

// Same fragment, built from f32 memory with on-the-fly bf16 conversion.
static __device__ inline FragB load_frag_f32cvt(const float* base, int row,
                                                int stride, int kb) {
    const float* p = base + (size_t)row * stride + kb;
    float4 a0 = *reinterpret_cast<const float4*>(p);
    float4 a1 = *reinterpret_cast<const float4*>(p + 4);
    float4 b0 = *reinterpret_cast<const float4*>(p + 16);
    float4 b1 = *reinterpret_cast<const float4*>(p + 20);
    FragB f;
    f.u[0] = pack2_bf16(a0.x, a0.y); f.u[1] = pack2_bf16(a0.z, a0.w);
    f.u[2] = pack2_bf16(a1.x, a1.y); f.u[3] = pack2_bf16(a1.z, a1.w);
    f.u[4] = pack2_bf16(b0.x, b0.y); f.u[5] = pack2_bf16(b0.z, b0.w);
    f.u[6] = pack2_bf16(b1.x, b1.y); f.u[7] = pack2_bf16(b1.z, b1.w);
    return f;
}

static __device__ inline float bnrelu(float v, float sc, float sh) {
    float t = fmaf(v, sc, sh);
    return t > 0.f ? t : 0.f;
}

// Fragment from f32 h with fused batchnorm(affine) + ReLU, then bf16 pack.
static __device__ inline FragB load_frag_bn(const float* hbase, int row, int kb,
                                            const float* sc, const float* sh) {
    const float* p = hbase + (size_t)row * OUT_FEA + kb;
    FragB f;
    #pragma unroll
    for (int g = 0; g < 2; ++g) {
        int ko = g * 16;
        float4 x0 = *reinterpret_cast<const float4*>(p + ko);
        float4 x1 = *reinterpret_cast<const float4*>(p + ko + 4);
        float e0 = bnrelu(x0.x, sc[kb+ko+0], sh[kb+ko+0]);
        float e1 = bnrelu(x0.y, sc[kb+ko+1], sh[kb+ko+1]);
        float e2 = bnrelu(x0.z, sc[kb+ko+2], sh[kb+ko+2]);
        float e3 = bnrelu(x0.w, sc[kb+ko+3], sh[kb+ko+3]);
        float e4 = bnrelu(x1.x, sc[kb+ko+4], sh[kb+ko+4]);
        float e5 = bnrelu(x1.y, sc[kb+ko+5], sh[kb+ko+5]);
        float e6 = bnrelu(x1.z, sc[kb+ko+6], sh[kb+ko+6]);
        float e7 = bnrelu(x1.w, sc[kb+ko+7], sh[kb+ko+7]);
        f.u[4*g+0] = pack2_bf16(e0, e1);
        f.u[4*g+1] = pack2_bf16(e2, e3);
        f.u[4*g+2] = pack2_bf16(e4, e5);
        f.u[4*g+3] = pack2_bf16(e6, e7);
    }
    return f;
}

// ---------------------------------------------------------------------------
// Kernel 0: transpose + f32->bf16 convert weights (W1t: 128x512, W2t: 128x128)
// ---------------------------------------------------------------------------
__global__ void k_cvt_w(const float* __restrict__ W1, const float* __restrict__ W2,
                        uint16_t* __restrict__ W1t, uint16_t* __restrict__ W2t) {
    int tid = blockIdx.x * 256 + threadIdx.x;
    if (tid < IN_FEA * OUT_FEA) {                 // W1: 512x128 -> W1t[n][k]
        int k = tid >> 7, n = tid & 127;
        W1t[n * IN_FEA + k] = f2bf(W1[k * OUT_FEA + n]);
    } else {
        int t = tid - IN_FEA * OUT_FEA;
        if (t < OUT_FEA * OUT_FEA) {              // W2: 128x128 -> W2t[n][k]
            int k = t >> 7, n = t & 127;
            W2t[n * OUT_FEA + k] = f2bf(W2[k * OUT_FEA + n]);
        }
    }
}

// ---------------------------------------------------------------------------
// Kernel 1: h = doc @ W1 + b1   (bf16 WMMA, f32 accumulate)
// W1t (128 KB) staged once into dynamic LDS by the Tensor Data Mover
// (tensor_load_to_lds + s_wait_tensorcnt).  One wave owns a 16-row tile;
// 16 K-chunks x 8 N-tiles of WMMA, B fragments from LDS.
// Also emits deterministic per-wave column partial sums / sumsq for BN stats.
// ---------------------------------------------------------------------------
__global__ void k_gemm1(const float* __restrict__ doc, const uint16_t* __restrict__ W1t,
                        const float* __restrict__ b1, float* __restrict__ h,
                        float* __restrict__ psum, float* __restrict__ psq) {
    extern __shared__ uint16_t s_w1t[];           // 128 x 512 bf16 = 128 KB

    if (threadIdx.x < 32) {                       // one TDM op from wave 0
        uint64_t ga = (uint64_t)(uintptr_t)W1t;
        uint32_t la = lds_off(s_w1t);
        // D# group0: count=1 | lds_addr | global_addr[56:0] | type=2
        u32x4 g0 = { 1u, la, (uint32_t)ga,
                     (uint32_t)((ga >> 32) & 0x01FFFFFFu) | 0x80000000u };
        // D# group1: data_size=2B; tensor 512x128, stride 512; tile 512x128
        i32x8 g1 = { (int)(1u << 16),             // workgroup_mask=0, data_size=1(2B)
                     (int)(512u << 16),           // tensor_dim0[15:0]=512
                     (int)(128u << 16),           // tensor_dim1[15:0]=128
                     (int)(512u << 16),           // tile_dim0=512
                     (int)128,                    // tile_dim1=128, tile_dim2=0
                     (int)512,                    // tensor_dim0_stride=512
                     0, 0 };
        i32x4 z4 = { 0, 0, 0, 0 };                // groups 2/3 unused (2D tensor)
        i32x8 z8 = { 0, 0, 0, 0, 0, 0, 0, 0 };
        __builtin_amdgcn_tensor_load_to_lds(g0, g1, z4, z4, z8, 0);
        __builtin_amdgcn_s_wait_tensorcnt(0);
    }
    __syncthreads();

    const int wave = blockIdx.x * 8 + (threadIdx.x >> 5);   // 0..1023
    const int lane = threadIdx.x & 31;
    const int m  = lane & 15;
    const int hh = lane >> 4;
    const int row = wave * 16 + m;

    AccF c[8];
    #pragma unroll
    for (int nt = 0; nt < 8; ++nt)
        #pragma unroll
        for (int r = 0; r < 8; ++r) c[nt].f[r] = 0.f;

    for (int kc = 0; kc < IN_FEA / 32; ++kc) {
        const int kb = hh * 8 + kc * 32;
        FragB a = load_frag_f32cvt(doc, row, IN_FEA, kb);
        #pragma unroll
        for (int nt = 0; nt < 8; ++nt) {
            FragB b = load_frag_bf16(s_w1t, nt * 16 + m, IN_FEA, kb);  // ds_load
            c[nt].v = wmma_bf16(a, b, c[nt].v);
        }
    }

    #pragma unroll
    for (int nt = 0; nt < 8; ++nt) {
        const int col = nt * 16 + m;
        const float bias = b1[col];
        float s = 0.f, q = 0.f;
        #pragma unroll
        for (int r = 0; r < 8; ++r) {
            float val = c[nt].f[r] + bias;             // D[r+8h][col]
            h[(size_t)(wave * 16 + r + 8 * hh) * OUT_FEA + col] = val;
            s += val; q += val * val;
        }
        s += __shfl_xor(s, 16, 32);                    // combine row-halves
        q += __shfl_xor(q, 16, 32);
        if (lane < 16) {
            psum[(size_t)wave * OUT_FEA + col] = s;
            psq [(size_t)wave * OUT_FEA + col] = q;
        }
    }
}

// ---------------------------------------------------------------------------
// Kernel 2: reduce BN partials -> fused scale/shift  (deterministic order)
// ---------------------------------------------------------------------------
__global__ void k_stats(const float* __restrict__ psum, const float* __restrict__ psq,
                        const float* __restrict__ gamma, const float* __restrict__ beta,
                        float* __restrict__ scale, float* __restrict__ shift) {
    const int col = threadIdx.x;                       // 128 threads
    float s = 0.f, q = 0.f;
    for (int i = 0; i < ROW_TILES_FULL; ++i) {
        s += psum[(size_t)i * OUT_FEA + col];
        q += psq [(size_t)i * OUT_FEA + col];
    }
    const float inv_n = 1.f / (float)ROWS;
    float mu  = s * inv_n;
    float var = fmaf(q, inv_n, -mu * mu);
    float sc  = gamma[col] * rsqrtf(var + 1e-5f);
    scale[col] = sc;
    shift[col] = fmaf(-mu, sc, beta[col]);
}

// ---------------------------------------------------------------------------
// Kernel 3: out = l2norm(relu(bn(h)) @ W2 + b2), de-interleaved to bf16
//           out1 = odd rows, out2 = even rows (each 8192x128).
// W2t (32 KB) staged to LDS via async-to-LDS.
// ---------------------------------------------------------------------------
__global__ void k_gemm2(const float* __restrict__ h, const uint16_t* __restrict__ W2t,
                        const float* __restrict__ b2,
                        const float* __restrict__ scale, const float* __restrict__ shift,
                        uint16_t* __restrict__ out1, uint16_t* __restrict__ out2) {
    __shared__ uint16_t s_w2t[OUT_FEA * OUT_FEA];      // 32 KB
    __shared__ float s_scale[OUT_FEA];
    __shared__ float s_shift[OUT_FEA];
    if (threadIdx.x < OUT_FEA) {
        s_scale[threadIdx.x] = scale[threadIdx.x];
        s_shift[threadIdx.x] = shift[threadIdx.x];
    }
    {   // 256 thr x 8 x 16B = 32 KB
        uint32_t base = lds_off(s_w2t);
        #pragma unroll
        for (int i = 0; i < 8; ++i) {
            int off = (threadIdx.x + i * 256) * 16;
            async_ld_b128(base + off, (const char*)W2t + off);
        }
        wait_async0();
    }
    __syncthreads();

    const int wave = blockIdx.x * 8 + (threadIdx.x >> 5);   // 0..1023
    const int lane = threadIdx.x & 31;
    const int m  = lane & 15;
    const int hh = lane >> 4;
    const int row = wave * 16 + m;

    AccF c[8];
    #pragma unroll
    for (int nt = 0; nt < 8; ++nt)
        #pragma unroll
        for (int r = 0; r < 8; ++r) c[nt].f[r] = 0.f;

    #pragma unroll
    for (int kc = 0; kc < OUT_FEA / 32; ++kc) {
        const int kb = hh * 8 + kc * 32;
        FragB a = load_frag_bn(h, row, kb, s_scale, s_shift);
        #pragma unroll
        for (int nt = 0; nt < 8; ++nt) {
            FragB b = load_frag_bf16(s_w2t, nt * 16 + m, OUT_FEA, kb);  // ds_load
            c[nt].v = wmma_bf16(a, b, c[nt].v);
        }
    }

    // + bias, per-row L2 norm, scale, de-interleave store
    #pragma unroll
    for (int nt = 0; nt < 8; ++nt) {
        const float bias = b2[nt * 16 + m];
        #pragma unroll
        for (int r = 0; r < 8; ++r) c[nt].f[r] += bias;
    }
    float inv[8];
    #pragma unroll
    for (int r = 0; r < 8; ++r) {
        float ss = 0.f;
        #pragma unroll
        for (int nt = 0; nt < 8; ++nt) ss = fmaf(c[nt].f[r], c[nt].f[r], ss);
        ss += __shfl_xor(ss, 1, 32);
        ss += __shfl_xor(ss, 2, 32);
        ss += __shfl_xor(ss, 4, 32);
        ss += __shfl_xor(ss, 8, 32);                    // full sum within 16-lane half
        inv[r] = 1.f / fmaxf(sqrtf(ss), 1e-12f);
    }
    #pragma unroll
    for (int r = 0; r < 8; ++r) {
        const int g = wave * 16 + r + 8 * hh;           // global row
        uint16_t* dst = ((g & 1) ? out1 : out2) + (size_t)(g >> 1) * OUT_FEA;
        #pragma unroll
        for (int nt = 0; nt < 8; ++nt)
            dst[nt * 16 + m] = f2bf(c[nt].f[r] * inv[r]);
    }
}

// ---------------------------------------------------------------------------
// Kernel 4: all four Gram row-sum passes in one launch (blockIdx.y selects
// the (X,Y) pair).  rs[i] = sum_j exp(2 * X[i].Y[j]) ; diag(1,2) -> pos.
// Per block: Y tiles (4 KB) double-buffered in LDS via async-to-LDS DMA,
// shared by 8 waves; each wave holds TWO 16-row A tiles in registers so each
// LDS B fragment feeds 2 WMMAs.  ~69 GFLOP of bf16 WMMA total.
// ---------------------------------------------------------------------------
__global__ void k_gram_all(const uint16_t* __restrict__ out1, const uint16_t* __restrict__ out2,
                           float* __restrict__ rs11, float* __restrict__ rs22,
                           float* __restrict__ rs12, float* __restrict__ rs21,
                           float* __restrict__ posp) {
    __shared__ uint16_t s_y[2][16 * OUT_FEA];          // 2 x 4 KB tiles

    const uint16_t* X; const uint16_t* Y; float* rs; float* pos = nullptr;
    switch (blockIdx.y) {
        case 0:  X = out1; Y = out1; rs = rs11; break;
        case 1:  X = out2; Y = out2; rs = rs22; break;
        case 2:  X = out1; Y = out2; rs = rs12; pos = posp; break;
        default: X = out2; Y = out1; rs = rs21; break;
    }

    const int wave = threadIdx.x >> 5;
    const int lane = threadIdx.x & 31;
    const int m  = lane & 15;
    const int hh = lane >> 4;
    const int it0 = (blockIdx.x * 8 + wave) * 2;       // two row tiles per wave
    const int it1 = it0 + 1;

    FragB a0[4], a1[4];
    #pragma unroll
    for (int kc = 0; kc < 4; ++kc) {
        a0[kc] = load_frag_bf16(X, it0 * 16 + m, OUT_FEA, hh * 8 + kc * 32);
        a1[kc] = load_frag_bf16(X, it1 * 16 + m, OUT_FEA, hh * 8 + kc * 32);
    }

    float acc0[8], acc1[8];
    #pragma unroll
    for (int r = 0; r < 8; ++r) { acc0[r] = 0.f; acc1[r] = 0.f; }

    // prime the pipeline: tile 0 -> buffer 0 (256 lanes x 16B = 4 KB)
    async_ld_b128(lds_off(&s_y[0][0]) + threadIdx.x * 16,
                  (const char*)Y + threadIdx.x * 16);

    for (int jt = 0; jt < ROW_TILES_HALF; ++jt) {
        if (jt + 1 < ROW_TILES_HALF) {                 // DMA next tile, overlap
            async_ld_b128(lds_off(&s_y[(jt + 1) & 1][0]) + threadIdx.x * 16,
                          (const char*)Y + (size_t)(jt + 1) * 4096 + threadIdx.x * 16);
            wait_async1();                             // tile jt complete (own ops)
        } else {
            wait_async0();
        }
        if (jt + 2 < ROW_TILES_HALF)                   // warm L2 ahead of DMA
            __builtin_prefetch((const char*)Y + (size_t)(jt + 2) * 4096 + threadIdx.x * 16, 0, 2);
        __syncthreads();                               // whole tile visible

        const uint16_t* tile = &s_y[jt & 1][0];
        AccF c0, c1;
        #pragma unroll
        for (int r = 0; r < 8; ++r) { c0.f[r] = 0.f; c1.f[r] = 0.f; }
        #pragma unroll
        for (int kc = 0; kc < 4; ++kc) {
            FragB b = load_frag_bf16(tile, m, OUT_FEA, hh * 8 + kc * 32);  // ds_load
            c0.v = wmma_bf16(a0[kc], b, c0.v);
            c1.v = wmma_bf16(a1[kc], b, c1.v);
        }

        const int d0 = (pos != nullptr) && (jt == it0);
        const int d1 = (pos != nullptr) && (jt == it1);
        #pragma unroll
        for (int r = 0; r < 8; ++r) {
            float e0 = __expf(2.0f * c0.f[r]);         // exp(dot / TEM), TEM=0.5
            float e1 = __expf(2.0f * c1.f[r]);
            acc0[r] += e0;
            acc1[r] += e1;
            if (d0 && (m - 8 * hh) == r) pos[it0 * 16 + m] = e0;
            if (d1 && (m - 8 * hh) == r) pos[it1 * 16 + m] = e1;
        }
        __syncthreads();                               // buffer reuse fence
    }

    #pragma unroll
    for (int r = 0; r < 8; ++r) {
        float s0 = acc0[r], s1 = acc1[r];
        s0 += __shfl_xor(s0, 1, 32);  s1 += __shfl_xor(s1, 1, 32);
        s0 += __shfl_xor(s0, 2, 32);  s1 += __shfl_xor(s1, 2, 32);
        s0 += __shfl_xor(s0, 4, 32);  s1 += __shfl_xor(s1, 4, 32);
        s0 += __shfl_xor(s0, 8, 32);  s1 += __shfl_xor(s1, 8, 32);
        if (m == 0) {
            rs[it0 * 16 + hh * 8 + r] = s0;            // rows r+8h of tile 0
            rs[it1 * 16 + hh * 8 + r] = s1;            // rows r+8h of tile 1
        }
    }
}

// ---------------------------------------------------------------------------
// Kernel 5: final loss scalar
//   li = 0.5*(log d1 + log d2) - log pos ;  d = rs_self + rs_cross - e^2
// ---------------------------------------------------------------------------
__global__ void k_loss(const float* __restrict__ rs11, const float* __restrict__ rs22,
                       const float* __restrict__ rs12, const float* __restrict__ rs21,
                       const float* __restrict__ pos, float* __restrict__ out) {
    __shared__ float red[256];
    float s = 0.f;
    for (int i = threadIdx.x; i < HALF_N; i += 256) {
        float d1 = rs11[i] + rs12[i] - E2;
        float d2 = rs22[i] + rs21[i] - E2;
        s += 0.5f * (__logf(d1) + __logf(d2)) - __logf(pos[i]);
    }
    red[threadIdx.x] = s;
    __syncthreads();
    for (int w = 128; w > 0; w >>= 1) {
        if (threadIdx.x < w) red[threadIdx.x] += red[threadIdx.x + w];
        __syncthreads();
    }
    if (threadIdx.x == 0) out[0] = red[0] / (float)HALF_N;
}

// ---------------------------------------------------------------------------
// Host-side orchestration
// ---------------------------------------------------------------------------
static inline size_t align_up(size_t x, size_t a) { return (x + a - 1) & ~(a - 1); }

extern "C" void kernel_launch(void* const* d_in, const int* in_sizes, int n_in,
                              void* d_out, int out_size, void* d_ws, size_t ws_size,
                              hipStream_t stream) {
    (void)in_sizes; (void)n_in; (void)out_size; (void)ws_size;
    const float* doc   = (const float*)d_in[0];   // 16384x512
    const float* W1    = (const float*)d_in[1];   // 512x128
    const float* b1    = (const float*)d_in[2];   // 128
    const float* gamma = (const float*)d_in[3];   // 128
    const float* beta  = (const float*)d_in[4];   // 128
    const float* W2    = (const float*)d_in[5];   // 128x128
    const float* b2    = (const float*)d_in[6];   // 128
    float* out = (float*)d_out;                   // 1 scalar

    // Workspace carve-up (~14 MB total)
    uint8_t* ws = (uint8_t*)d_ws;
    size_t o = 0;
    uint16_t* W1t  = (uint16_t*)(ws + o); o = align_up(o + (size_t)IN_FEA * OUT_FEA * 2, 512);
    uint16_t* W2t  = (uint16_t*)(ws + o); o = align_up(o + (size_t)OUT_FEA * OUT_FEA * 2, 512);
    float*    h    = (float*)(ws + o);    o = align_up(o + (size_t)ROWS * OUT_FEA * 4, 512);
    float*    psum = (float*)(ws + o);    o = align_up(o + (size_t)ROW_TILES_FULL * OUT_FEA * 4, 512);
    float*    psq  = (float*)(ws + o);    o = align_up(o + (size_t)ROW_TILES_FULL * OUT_FEA * 4, 512);
    float*    scl  = (float*)(ws + o);    o = align_up(o + OUT_FEA * 4, 512);
    float*    shf  = (float*)(ws + o);    o = align_up(o + OUT_FEA * 4, 512);
    uint16_t* out1 = (uint16_t*)(ws + o); o = align_up(o + (size_t)HALF_N * OUT_FEA * 2, 512);
    uint16_t* out2 = (uint16_t*)(ws + o); o = align_up(o + (size_t)HALF_N * OUT_FEA * 2, 512);
    float*    rs11 = (float*)(ws + o);    o = align_up(o + HALF_N * 4, 512);
    float*    rs22 = (float*)(ws + o);    o = align_up(o + HALF_N * 4, 512);
    float*    rs12 = (float*)(ws + o);    o = align_up(o + HALF_N * 4, 512);
    float*    rs21 = (float*)(ws + o);    o = align_up(o + HALF_N * 4, 512);
    float*    pos  = (float*)(ws + o);    o = align_up(o + HALF_N * 4, 512);

    // 0) weight transpose + bf16 convert
    k_cvt_w<<<(IN_FEA * OUT_FEA + OUT_FEA * OUT_FEA + 255) / 256, 256, 0, stream>>>(
        W1, W2, W1t, W2t);

    // 1) GEMM1 + BN partial stats; W1t lives in 128 KB dynamic LDS (TDM-loaded)
    k_gemm1<<<ROW_TILES_FULL / 8, 256, 128 * 1024, stream>>>(doc, W1t, b1, h, psum, psq);

    // 2) BN stats -> fused scale/shift
    k_stats<<<1, OUT_FEA, 0, stream>>>(psum, psq, gamma, beta, scl, shf);

    // 3) BN+ReLU+GEMM2+L2norm, de-interleave to bf16 out1/out2
    k_gemm2<<<ROW_TILES_FULL / 8, 256, 0, stream>>>(h, W2t, b2, scl, shf, out1, out2);

    // 4) all four Gram row-sum passes in one grid (y = pair selector)
    k_gram_all<<<dim3(ROW_TILES_HALF / 16, 4, 1), 256, 0, stream>>>(
        out1, out2, rs11, rs22, rs12, rs21, pos);

    // 5) reduce to scalar loss
    k_loss<<<1, 256, 0, stream>>>(rs11, rs22, rs12, rs21, pos, out);
}
